// PlainPointTransformer_5093831213092
// MI455X (gfx1250) — compile-verified, matrix operand order checked by reading, not run on hardware
//
#include <hip/hip_runtime.h>

typedef __attribute__((ext_vector_type(16))) __bf16 v16bf;
typedef __attribute__((ext_vector_type(8)))  __bf16 v8bf;
typedef __attribute__((ext_vector_type(8)))  float  v8f;

#define NPTS 65536
#define CCH  256

static __device__ inline __bf16 f2bf(float f) {
    unsigned u = __builtin_bit_cast(unsigned, f);
    unsigned r = u + 0x7FFFu + ((u >> 16) & 1u);   // round-to-nearest-even
    unsigned short h = (unsigned short)(r >> 16);
    return __builtin_bit_cast(__bf16, h);
}

static __device__ inline float bf2f(__bf16 h) {
    unsigned short s = __builtin_bit_cast(unsigned short, h);
    unsigned u = (unsigned)s << 16;
    return __builtin_bit_cast(float, u);
}

static __device__ inline v16bf frag16(const __bf16* p0, const __bf16* p1) {
    v16bf r;
    ((v8bf*)&r)[0] = *(const v8bf*)p0;
    ((v8bf*)&r)[1] = *(const v8bf*)p1;
    return r;
}

// ---------------------------------------------------------------------------
// Weight packing: fp32 row-major W[K][Mout] -> bf16 WMMA-B fragment panels.
// Panel = one 32x16 (KxN) tile: 32 lanes x 16 bf16, in v_wmma_*_bf16 B layout
// (lane l holds column n = l&15; K-half select = l>>4; element i covers
//  K = 8*half+i (i<8) else 16+8*half+(i-8)).  Panels indexed [nt][kp].
// ---------------------------------------------------------------------------
__global__ __launch_bounds__(256) void pack_w(const float* __restrict__ W,
                                              __bf16* __restrict__ dst,
                                              int K, int Mout) {
    size_t tid = (size_t)blockIdx.x * 256 + threadIdx.x;
    if (tid >= (size_t)K * Mout) return;
    int i    = (int)(tid & 15);
    int lane = (int)((tid >> 4) & 31);
    size_t p = tid >> 9;                 // panel index
    int kpc  = K >> 5;
    int nt   = (int)(p / kpc);
    int kp   = (int)(p % kpc);
    int n    = nt * 16 + (lane & 15);
    int half = lane >> 4;
    int k    = kp * 32 + ((i < 8) ? (8 * half + i) : (16 + 8 * half + (i - 8)));
    dst[tid] = f2bf(W[(size_t)k * Mout + n]);
}

// ---------------------------------------------------------------------------
// WMMA GEMM: C[M x Mout] = act( A[M x K] * Bpack + bias ) (+ residual)
// block = 256 threads = 8 waves; wave -> 16 rows x 64 cols (4 wmma tiles).
// grid = (M/128, Mout/64).  All 8 waves share the same 64 output columns, so
// the block's B panels (4*kpc panels, contiguous in Bpack) are staged once
// into LDS; WMMA B-fragments come from LDS (ds_load_b128), A from global.
// K-loop is manually double-buffered: two kp-steps of fragments (2 A frags +
// 8 B frags in distinct registers) are loaded before the 8 WMMAs issue, so
// set-1 loads overlap set-0 XDL work.
// Dynamic LDS bytes = (K/32)*4096  (K=256 -> 32KB, K=1024 -> 128KB).
// ---------------------------------------------------------------------------
template<bool BF16OUT, bool RES, bool GELU_, bool BIAS>
__global__ __launch_bounds__(256) void wmma_gemm(const __bf16* __restrict__ A,
                                                 const __bf16* __restrict__ Bp,
                                                 const float* __restrict__ bias,
                                                 void* __restrict__ Cout,
                                                 int K, int Mout) {
    extern __shared__ char smem[];
    __bf16* Bs = (__bf16*)smem;

    const int lane    = threadIdx.x & 31;
    const int wave    = threadIdx.x >> 5;
    const int half    = lane >> 4;
    const int lrow    = lane & 15;
    const int rowbase = blockIdx.x * 128 + wave * 16;
    const int colbase = blockIdx.y * 64;
    const int kpc     = K >> 5;

    // ---- stage this block's B panels into LDS (one contiguous copy) ----
    {
        const uint4* src = (const uint4*)(Bp + (((size_t)(colbase >> 4) * kpc) << 9));
        uint4* dst = (uint4*)smem;
        const int chunks = kpc * 256;          // 4*kpc*1024 bytes / 16
        for (int c = threadIdx.x; c < chunks; c += 256) dst[c] = src[c];
    }
    __syncthreads();

    const __bf16* Arow = A + (size_t)(rowbase + lrow) * K + 8 * half;
    const __bf16* Bl   = Bs + lane * 16;

    v8f acc[4] = {};
    for (int kp = 0; kp < kpc; kp += 2) {       // kpc is 8 or 32 (even)
        const __bf16* ab0 = Arow + kp * 32;
        v16bf a0 = frag16(ab0, ab0 + 16);
        v16bf a1 = frag16(ab0 + 32, ab0 + 48);

        v16bf b0[4], b1[4];
#pragma unroll
        for (int t = 0; t < 4; ++t) {
            const __bf16* pb = Bl + ((size_t)(t * kpc + kp) << 9);
            b0[t] = frag16(pb, pb + 8);
            b1[t] = frag16(pb + 512, pb + 520);
        }
#pragma unroll
        for (int t = 0; t < 4; ++t) {
            acc[t] = __builtin_amdgcn_wmma_f32_16x16x32_bf16(
                false, a0, false, b0[t], (short)0, acc[t], false, false);
        }
#pragma unroll
        for (int t = 0; t < 4; ++t) {
            acc[t] = __builtin_amdgcn_wmma_f32_16x16x32_bf16(
                false, a1, false, b1[t], (short)0, acc[t], false, false);
        }
    }

#pragma unroll
    for (int t = 0; t < 4; ++t) {
        int n = colbase + t * 16 + lrow;
        float bs = BIAS ? bias[n] : 0.0f;
#pragma unroll
        for (int r = 0; r < 8; ++r) {
            int m = rowbase + r + 8 * half;
            float val = acc[t][r] + bs;
            if (GELU_) val = 0.5f * val * (1.0f + erff(val * 0.70710678118f));
            size_t off = (size_t)m * Mout + n;
            if (BF16OUT) {
                ((__bf16*)Cout)[off] = f2bf(val);
            } else {
                float* o = (float*)Cout;
                o[off] = RES ? (o[off] + val) : val;
            }
        }
    }
}

// ---------------------------------------------------------------------------
// LayerNorm: one wave per row of 256 channels (8 per lane), bf16 output.
// ---------------------------------------------------------------------------
__global__ __launch_bounds__(256) void ln_rows(const float* __restrict__ x,
                                               const float* __restrict__ g,
                                               const float* __restrict__ b,
                                               __bf16* __restrict__ out) {
    int wave = threadIdx.x >> 5, lane = threadIdx.x & 31;
    int row  = blockIdx.x * 8 + wave;
    const float* xr = x + (size_t)row * CCH;
    float vals[8], s = 0.0f, s2 = 0.0f;
#pragma unroll
    for (int j = 0; j < 8; ++j) {
        float v = xr[lane * 8 + j];
        vals[j] = v; s += v; s2 += v * v;
    }
#pragma unroll
    for (int off = 16; off > 0; off >>= 1) {
        s  += __shfl_xor(s,  off, 32);
        s2 += __shfl_xor(s2, off, 32);
    }
    float mean = s * (1.0f / CCH);
    float var  = s2 * (1.0f / CCH) - mean * mean;
    float inv  = rsqrtf(var + 1e-5f);
    v8bf ov;
#pragma unroll
    for (int j = 0; j < 8; ++j) {
        int c = lane * 8 + j;
        ov[j] = f2bf((vals[j] - mean) * inv * g[c] + b[c]);
    }
    *(v8bf*)(out + (size_t)row * CCH + lane * 8) = ov;
}

// ---------------------------------------------------------------------------
// KNN attention: one wave per point. qkv row = [q(256) | k(256) | v(256)].
// scores_j = (q . k[idx_j]) / 16, softmax over 16, out = sum attn_j * v[idx_j]
// ---------------------------------------------------------------------------
__global__ __launch_bounds__(256) void knn_attn(const __bf16* __restrict__ qkv,
                                                const int* __restrict__ knn,
                                                __bf16* __restrict__ out) {
    int wave = threadIdx.x >> 5, lane = threadIdx.x & 31;
    int n = blockIdx.x * 8 + wave;
    const __bf16* qr = qkv + (size_t)n * 768 + lane * 8;
    float qv[8];
    {
        v8bf q8 = *(const v8bf*)qr;
#pragma unroll
        for (int t = 0; t < 8; ++t) qv[t] = bf2f(q8[t]);
    }
    const int* idx = knn + (size_t)n * 16;
    int nb[16];
#pragma unroll
    for (int j = 0; j < 16; ++j) nb[j] = idx[j];

    float scores[16];
#pragma unroll
    for (int j = 0; j < 16; ++j) {
        v8bf k8 = *(const v8bf*)(qkv + (size_t)nb[j] * 768 + 256 + lane * 8);
        float d = 0.0f;
#pragma unroll
        for (int t = 0; t < 8; ++t) d += qv[t] * bf2f(k8[t]);
#pragma unroll
        for (int off = 16; off > 0; off >>= 1) d += __shfl_xor(d, off, 32);
        scores[j] = d * 0.0625f;   // C^-0.5 = 1/16
    }
    float mx = scores[0];
#pragma unroll
    for (int j = 1; j < 16; ++j) mx = fmaxf(mx, scores[j]);
    float sum = 0.0f;
#pragma unroll
    for (int j = 0; j < 16; ++j) { scores[j] = __expf(scores[j] - mx); sum += scores[j]; }
    float rinv = 1.0f / sum;

    float o[8] = {0, 0, 0, 0, 0, 0, 0, 0};
#pragma unroll
    for (int j = 0; j < 16; ++j) {
        v8bf v8 = *(const v8bf*)(qkv + (size_t)nb[j] * 768 + 512 + lane * 8);
        float a = scores[j] * rinv;
#pragma unroll
        for (int t = 0; t < 8; ++t) o[t] += a * bf2f(v8[t]);
    }
    v8bf ov;
#pragma unroll
    for (int t = 0; t < 8; ++t) ov[t] = f2bf(o[t]);
    *(v8bf*)(out + (size_t)n * CCH + lane * 8) = ov;
}

// ---------------------------------------------------------------------------
extern "C" void kernel_launch(void* const* d_in, const int* in_sizes, int n_in,
                              void* d_out, int out_size, void* d_ws, size_t ws_size,
                              hipStream_t stream) {
    (void)in_sizes; (void)n_in; (void)out_size; (void)ws_size;

    const float* x_in   = (const float*)d_in[0];
    const int*   knn    = (const int*)d_in[1];
    const float* ln1_g  = (const float*)d_in[2];
    const float* ln1_b  = (const float*)d_in[3];
    const float* w_qkv  = (const float*)d_in[4];
    const float* w_proj = (const float*)d_in[5];
    const float* b_proj = (const float*)d_in[6];
    const float* ln2_g  = (const float*)d_in[7];
    const float* ln2_b  = (const float*)d_in[8];
    const float* w_fc1  = (const float*)d_in[9];
    const float* b_fc1  = (const float*)d_in[10];
    const float* w_fc2  = (const float*)d_in[11];
    const float* b_fc2  = (const float*)d_in[12];
    float* x = (float*)d_out;

    // workspace layout (bytes):
    //   [0, 6.29M)   packed bf16 weights (4 layers x 786432 elems)
    //   [+33.5M)     hbuf: LN output, N x 256 bf16
    //   [+134.2M)    bufA: qkv (N x 768 bf16) ++ attnout (N x 256 bf16),
    //                reused as fc1 output (N x 1024 bf16)
    char* ws = (char*)d_ws;
    __bf16* Wpack = (__bf16*)ws;
    __bf16* hbuf  = (__bf16*)(ws + 6291456);
    __bf16* qkvb  = (__bf16*)(ws + 6291456 + 33554432);
    __bf16* attno = (__bf16*)(ws + 6291456 + 33554432 + 100663296);
    __bf16* h2buf = qkvb;   // fc1 output reuses qkv region (dead by then)

    const size_t LSTRIDE = 786432;
    const size_t OFF_QKV = 0, OFF_PROJ = 196608, OFF_FC1 = 262144, OFF_FC2 = 524288;

    // pack all weights (runs every call; deterministic)
    for (int i = 0; i < 4; ++i) {
        size_t base = (size_t)i * LSTRIDE;
        pack_w<<<196608 / 256, 256, 0, stream>>>(w_qkv  + (size_t)i * 256 * 768,  Wpack + base + OFF_QKV,  256,  768);
        pack_w<<< 65536 / 256, 256, 0, stream>>>(w_proj + (size_t)i * 256 * 256,  Wpack + base + OFF_PROJ, 256,  256);
        pack_w<<<262144 / 256, 256, 0, stream>>>(w_fc1  + (size_t)i * 256 * 1024, Wpack + base + OFF_FC1,  256, 1024);
        pack_w<<<262144 / 256, 256, 0, stream>>>(w_fc2  + (size_t)i * 1024 * 256, Wpack + base + OFF_FC2, 1024,  256);
    }

    hipMemcpyAsync(x, x_in, (size_t)NPTS * CCH * sizeof(float),
                   hipMemcpyDeviceToDevice, stream);

    const size_t LDS_K256  = (256 >> 5) * 4096;    // 32 KB
    const size_t LDS_K1024 = (1024 >> 5) * 4096;   // 128 KB

    for (int i = 0; i < 4; ++i) {
        size_t base = (size_t)i * LSTRIDE;
        // attention branch
        ln_rows<<<NPTS / 8, 256, 0, stream>>>(x, ln1_g + i * CCH, ln1_b + i * CCH, hbuf);
        wmma_gemm<true, false, false, false>
            <<<dim3(NPTS / 128, 768 / 64), 256, LDS_K256, stream>>>(hbuf, Wpack + base + OFF_QKV, nullptr, qkvb, 256, 768);
        knn_attn<<<NPTS / 8, 256, 0, stream>>>(qkvb, knn, attno);
        wmma_gemm<false, true, false, true>
            <<<dim3(NPTS / 128, 256 / 64), 256, LDS_K256, stream>>>(attno, Wpack + base + OFF_PROJ, b_proj + i * CCH, x, 256, 256);
        // MLP branch
        ln_rows<<<NPTS / 8, 256, 0, stream>>>(x, ln2_g + i * CCH, ln2_b + i * CCH, hbuf);
        wmma_gemm<true, false, true, true>
            <<<dim3(NPTS / 128, 1024 / 64), 256, LDS_K256, stream>>>(hbuf, Wpack + base + OFF_FC1, b_fc1 + i * 1024, h2buf, 256, 1024);
        wmma_gemm<false, true, false, true>
            <<<dim3(NPTS / 128, 256 / 64), 256, LDS_K1024, stream>>>(h2buf, Wpack + base + OFF_FC2, b_fc2 + i * CCH, x, 1024, 256);
    }
}
